// KnowledgeBase_35596688949601
// MI455X (gfx1250) — matrix operand order repaired and checked by reading, not compiled, not run on hardware
//
#include <hip/hip_runtime.h>
#include <stdint.h>

typedef __attribute__((ext_vector_type(16))) _Float16 v16h;
typedef __attribute__((ext_vector_type(8)))  float    v8f;

union Frag { v16h v; uint32_t u[8]; };

__device__ __forceinline__ uint16_t h_bits(_Float16 h) {
    union { _Float16 h; uint16_t u; } cv; cv.h = h; return cv.u;
}

__global__ __launch_bounds__(256) void kb_wmma_kernel(
    const float* __restrict__ query,
    const _Float16* __restrict__ storage,
    float* __restrict__ out,
    int N)
{
    // per wave: 32 query rows x 32 f16 weights (16 dwords) = 2KB; 8 waves = 16KB
    __shared__ uint32_t smem[8 * 512];

    const int lane = threadIdx.x & 31;
    const int col  = lane & 15;          // N-column / M-row selector within half-wave
    const int hi16 = (lane >> 4) & 1;    // upper 16 lanes flag

    // wave index as an SGPR -> all loop bookkeeping scalar, uniform branches
    const int wave  = __builtin_amdgcn_readfirstlane(threadIdx.x >> 5);
    uint32_t* wrow  = &smem[wave * 512];

    // ---- one-time: build 4 resident B fragments (32x16 f16) from storage rows 0..31.
    // B k order used here (contiguous k, same as A): lanes 0-15 hold K=0..15, lanes 16-31 hold K=16..31,
    // VGPR i packs K=2i (lo), 2i+1 (hi); column N = c*16 + (lane&15).
    const uint16_t* sb = (const uint16_t*)storage;
    Frag bfrag[4];
    #pragma unroll
    for (int c = 0; c < 4; ++c) {
        int n = c * 16 + col;
        #pragma unroll
        for (int i = 0; i < 8; ++i) {
            int k0 = 2 * i + hi16 * 16;
            uint32_t lo = sb[(size_t)k0 * 64 + n];
            uint32_t hi = sb[(size_t)(k0 + 1) * 64 + n];
            bfrag[c].u[i] = lo | (hi << 16);
        }
    }

    const int units = (N + 31) >> 5;
    const int wgrid = gridDim.x * (blockDim.x >> 5);        // scalar
    const int gwave = blockIdx.x * (blockDim.x >> 5) + wave; // scalar

    for (int unit = gwave; unit < units; unit += wgrid) {
        const int qbase = unit << 5;                 // scalar
        const bool full = (qbase + 32) <= N;         // scalar -> s_cbranch

        // ---- per-lane: weights for query qbase+lane -----------------------
        int q  = qbase + lane;
        int qc = q < N ? q : N - 1;
        float4 qv = reinterpret_cast<const float4*>(query)[qc];

        float s0 = (qv.x - floorf(qv.x)) * 32.f;
        float s1 = (qv.y - floorf(qv.y)) * 32.f;
        float s2 = (qv.z - floorf(qv.z)) * 32.f;
        float s3 = (qv.w - floorf(qv.w)) * 32.f;
        float a3 = rintf(s3);
        float d0 = s0 - rintf(s0), d1 = s1 - rintf(s1), d2 = s2 - rintf(s2), d3 = s3 - a3;
        float t0 = fabsf(d0), t1 = fabsf(d1), t2 = fabsf(d2), t3 = fabsf(d3);

        float w4 = 4.f - (t0 + t1 + t2 + t3);                 // anchor-point weight
        float w0 = (d0 != 0.f) ? (w4 - 1.f + 2.f * t0) : w4;
        float w1 = (d1 != 0.f) ? (w4 - 1.f + 2.f * t1) : w4;
        float w2 = (d2 != 0.f) ? (w4 - 1.f + 2.f * t2) : w4;
        float w3 = (d3 != 0.f) ? (w4 - 1.f + 2.f * t3) : w4;

        // mod-32 collapses all but the last coordinate of the hash
        int ia3 = (int)a3;
        int fa  = ia3 & 31;
        int st3 = (d3 > 0.f) ? 1 : ((d3 < 0.f) ? -1 : 0);
        int fb  = (ia3 + st3) & 31;                           // floor-mod for [-1,33]

        float wa = (w0 + w1 + w2 + w4) * 0.25f;               // fold the /Q in
        float wb = w3 * 0.25f;

        // ---- sparse scatter: zero 64B row, then write the <=2 nonzero f16 --
        uint16_t* myrow16 = (uint16_t*)&wrow[lane * 16];
        {
            uint4 z = make_uint4(0u, 0u, 0u, 0u);
            uint4* zp = (uint4*)myrow16;
            zp[0] = z; zp[1] = z; zp[2] = z; zp[3] = z;       // 4x ds_store_b128

            // collision (fb==fa, i.e. d3==0) pre-summed into first write
            _Float16 va = (fa == fb) ? (_Float16)(wa + wb) : (_Float16)wa;
            myrow16[fa] = h_bits(va);                         // ds_store_b16
            if (fb != fa)
                myrow16[fb] = h_bits((_Float16)wb);           // ds_store_b16
        }

        __asm__ volatile("s_wait_dscnt 0" ::: "memory");   // intra-wave LDS RAW

        // one 64-bit address computation per unit; all stores use imm offsets
        float* pout = out + (size_t)qbase * 64 + (size_t)(hi16 * 8) * 64 + col;

        // ---- 2 M-tiles x 4 N-tiles of V_WMMA_F32_16X16X32_F16 -------------
        #pragma unroll
        for (int mt = 0; mt < 2; ++mt) {
            // A as loaded here (contiguous k, same as B): lane row M = lane&15;
            // lanes 0-15: K=0-15  -> row dwords 0-7
            // lanes16-31: K=16-31 -> row dwords 8-15
            const uint32_t* arow = &wrow[(mt * 16 + col) * 16];
            Frag af;
            const int b0 = hi16 * 8;
            const int b1 = 4 + hi16 * 8;
            #pragma unroll
            for (int i = 0; i < 4; ++i) {
                af.u[i]     = arow[b0 + i];
                af.u[4 + i] = arow[b1 + i];
            }

            #pragma unroll
            for (int c = 0; c < 4; ++c) {
                v8f acc = {};
                acc = __builtin_amdgcn_wmma_f32_16x16x32_f16(
                        false, af.v, false, bfrag[c].v,
                        (short)0, acc, false, false);

                // D layout: VGPR v -> M = v + hi16*8; column = lane&15
                float* p = pout + mt * 1024 + c * 16;   // constant-folded offsets
                if (full) {
                    #pragma unroll
                    for (int v = 0; v < 8; ++v)
                        p[v * 64] = acc[v];             // clause of global_store_b32
                } else {
                    const int rbase = qbase + mt * 16 + hi16 * 8;
                    #pragma unroll
                    for (int v = 0; v < 8; ++v)
                        if (rbase + v < N)
                            p[v * 64] = acc[v];
                }
            }
        }
    }
}

extern "C" void kernel_launch(void* const* d_in, const int* in_sizes, int n_in,
                              void* d_out, int out_size, void* d_ws, size_t ws_size,
                              hipStream_t stream) {
    (void)n_in; (void)out_size; (void)d_ws; (void)ws_size;
    const float*    query   = (const float*)d_in[0];
    const _Float16* storage = (const _Float16*)d_in[1];
    float*          out     = (float*)d_out;

    const int N = in_sizes[0] / 4;                 // queries (N,4) f32
    if (N <= 0) return;

    const int units  = (N + 31) / 32;              // 32 queries per wave-unit
    int blocks = (units + 7) / 8;                  // 8 waves per 256-thread block
    if (blocks > 8192) blocks = 8192;              // grid-stride covers the rest
    if (blocks < 1)    blocks = 1;

    kb_wmma_kernel<<<blocks, 256, 0, stream>>>(query, storage, out, N);
}
